// ReaderRankerClassifier_73856257622111
// MI455X (gfx1250) — matrix operand order repaired and checked by reading, not compiled, not run on hardware
//
#include <hip/hip_runtime.h>
#include <hip/hip_bf16.h>
#include <math.h>

// ---------------- problem constants ----------------
#define B_    4
#define NP_   10
#define PLEN_ 400
#define QLEN_ 50
#define EMB_  1024
#define H_    256
#define NH_   8
#define DFF_  1024
#define SCALE_C 0.17677669529663687f   // 1/sqrt(32)

#define RP_ (B_*NP_*PLEN_)   // 16000 passage rows
#define RQ_ (B_*QLEN_)       // 200 query rows

typedef __attribute__((ext_vector_type(16))) _Float16 v16h;
typedef __attribute__((ext_vector_type(8)))  float    v8f;
typedef __attribute__((ext_vector_type(4)))  float    f32x4;

__device__ __forceinline__ float sigmoidf_(float x){ return 1.0f/(1.0f+__expf(-x)); }
__device__ __forceinline__ float geluf_(float x){ return 0.5f*x*(1.0f+erff(x*0.70710678118654752f)); }

// =====================================================================
// WMMA GEMM:  C(M,N) = act( op(A) * op(B) + bias ), batched over gridDim.z
//  TA==0 : A is MxK row-major (lda)      TA==1 : A stored KxM row-major (A^T)
//  TB==0 : B is KxN row-major (ldb)      TB==1 : B stored NxK row-major (x@W.T)
//  ACT: 0 none, 1 relu, 2 sigmoid, 3 gelu(exact)
// One wave computes a 16x64 strip: 1 A-fragment reused across 4 B-fragments,
// 4 independent v_wmma_f32_16x16x32_f16 accumulator chains per k-step.
// Full k-steps use unconditional (vectorized) loads; single guarded tail uses
// address-clamp + value-select (no EXEC divergence).
// =====================================================================
__device__ __forceinline__ v16h cvt16_4x4(f32x4 q0, f32x4 q1, f32x4 q2, f32x4 q3)
{
  v16h b;
  #pragma unroll
  for (int i = 0; i < 4; ++i) {
    b[i]    = (_Float16)q0[i];  b[4+i]  = (_Float16)q1[i];
    b[8+i]  = (_Float16)q2[i];  b[12+i] = (_Float16)q3[i];
  }
  return b;
}

template<int TA, int TB, int ACT>
__global__ __launch_bounds__(128)
void gemm_wmma(const float* __restrict__ A0, const float* __restrict__ B0,
               const float* __restrict__ bias, float* __restrict__ C0,
               int M, int N, int K, int lda, int ldb, int ldc,
               size_t sA, size_t sB, size_t sC, int bdiv)
{
  const float* A  = A0 + (size_t)blockIdx.z * sA;
  const float* Bm = B0 + (size_t)(blockIdx.z / bdiv) * sB;
  float*       C  = C0 + (size_t)blockIdx.z * sC;

  const int lane = threadIdx.x & 31;
  const int wv   = threadIdx.x >> 5;       // 0..3
  const int tm   = blockIdx.y * 4 + wv;    // 16-row tile per wave
  const int tn   = blockIdx.x;             // 64-col block
  const int half = lane >> 4;              // 0/1
  const int l16  = lane & 15;

  int mrow = tm*16 + l16; if (mrow > M-1) mrow = M-1;   // clamp; stores guarded
  int nc[4];
  #pragma unroll
  for (int j = 0; j < 4; ++j) {
    int n = tn*64 + j*16 + l16;
    nc[j] = (n > N-1) ? (N-1) : n;
  }
  const int kbA = half * 8;     // A frag K-striping per ISA 7.12.2
  const int kbB = half * 8;     // B frag K-striping (same as A)

  v8f acc[4] = {};
  const int Kfull = K & ~31;
  int k0 = 0;

  for (; k0 < Kfull; k0 += 32) {
    v16h a;
    if (TA == 0) {
      const float* ap = A + (size_t)mrow * lda + k0 + kbA;
      if ((lda & 3) == 0) {
        f32x4 x0 = *(const f32x4*)(ap);
        f32x4 x1 = *(const f32x4*)(ap + 4);
        f32x4 y0 = *(const f32x4*)(ap + 16);
        f32x4 y1 = *(const f32x4*)(ap + 20);
        a = cvt16_4x4(x0, x1, y0, y1);
      } else {
        #pragma unroll
        for (int i = 0; i < 8; ++i) {
          a[i]   = (_Float16)ap[i];
          a[8+i] = (_Float16)ap[16 + i];
        }
      }
    } else {
      #pragma unroll
      for (int i = 0; i < 8; ++i) {
        a[i]   = (_Float16)A[(size_t)(k0 + kbA + i) * lda + mrow];
        a[8+i] = (_Float16)A[(size_t)(k0 + kbA + 16 + i) * lda + mrow];
      }
    }

    v16h bf[4];
    if (TB == 1) {
      if ((ldb & 3) == 0) {
        #pragma unroll
        for (int j = 0; j < 4; ++j) {
          const float* bp = Bm + (size_t)nc[j] * ldb + k0 + kbB;
          bf[j] = cvt16_4x4(*(const f32x4*)bp, *(const f32x4*)(bp + 4),
                            *(const f32x4*)(bp + 16), *(const f32x4*)(bp + 20));
        }
      } else {
        #pragma unroll
        for (int j = 0; j < 4; ++j) {
          const float* bp = Bm + (size_t)nc[j] * ldb + k0 + kbB;
          #pragma unroll
          for (int i = 0; i < 16; ++i) bf[j][i] = (_Float16)bp[i + (i & 8)];
        }
      }
    } else {
      #pragma unroll
      for (int j = 0; j < 4; ++j) {
        #pragma unroll
        for (int i = 0; i < 16; ++i)
          bf[j][i] = (_Float16)Bm[(size_t)(k0 + kbB + i + (i & 8)) * ldb + nc[j]];
      }
    }

    #pragma unroll
    for (int j = 0; j < 4; ++j)
      acc[j] = __builtin_amdgcn_wmma_f32_16x16x32_f16(false, a, false, bf[j],
                                                      (short)0, acc[j], false, false);
  }

  if (k0 < K) {   // guarded tail: clamp addresses, select values (no EXEC branches)
    v16h a;
    #pragma unroll
    for (int i = 0; i < 8; ++i) {
      int k1 = k0 + kbA + i;        int k1c = (k1 < K) ? k1 : (K - 1);
      int k2 = k0 + kbA + 16 + i;   int k2c = (k2 < K) ? k2 : (K - 1);
      float v1 = (TA == 0) ? A[(size_t)mrow * lda + k1c] : A[(size_t)k1c * lda + mrow];
      float v2 = (TA == 0) ? A[(size_t)mrow * lda + k2c] : A[(size_t)k2c * lda + mrow];
      a[i]   = (_Float16)((k1 < K) ? v1 : 0.0f);
      a[8+i] = (_Float16)((k2 < K) ? v2 : 0.0f);
    }
    v16h bf[4];
    #pragma unroll
    for (int j = 0; j < 4; ++j) {
      #pragma unroll
      for (int i = 0; i < 16; ++i) {
        int k1 = k0 + kbB + i + (i & 8);      int k1c = (k1 < K) ? k1 : (K - 1);
        float w = (TB == 1) ? Bm[(size_t)nc[j] * ldb + k1c]
                            : Bm[(size_t)k1c * ldb + nc[j]];
        bf[j][i] = (_Float16)((k1 < K) ? w : 0.0f);
      }
    }
    #pragma unroll
    for (int j = 0; j < 4; ++j)
      acc[j] = __builtin_amdgcn_wmma_f32_16x16x32_f16(false, a, false, bf[j],
                                                      (short)0, acc[j], false, false);
  }

  // ---- epilogue: VGPR r holds (m = 16*tm + r + 8*half, n)
  #pragma unroll
  for (int j = 0; j < 4; ++j) {
    int no = tn*64 + j*16 + l16;
    #pragma unroll
    for (int r = 0; r < 8; ++r) {
      int mo = tm*16 + r + half*8;
      if (mo < M && no < N) {
        float v = acc[j][r];
        if (bias) v += bias[no];
        if (ACT == 1) v = fmaxf(v, 0.0f);
        else if (ACT == 2) v = sigmoidf_(v);
        else if (ACT == 3) v = geluf_(v);
        C[(size_t)mo * ldc + no] = v;
      }
    }
  }
}

template<int TA, int TB, int ACT>
static void launch_gemm(const float* A, const float* Bm, const float* bias, float* C,
                        int M, int N, int K, int lda, int ldb, int ldc, hipStream_t s,
                        int batch = 1, size_t sA = 0, size_t sB = 0, size_t sC = 0,
                        int bdiv = 1)
{
  dim3 grid((N + 63) / 64, (M + 63) / 64, batch);
  gemm_wmma<TA, TB, ACT><<<grid, 128, 0, s>>>(A, Bm, bias, C, M, N, K, lda, ldb, ldc,
                                              sA, sB, sC, bdiv);
}

// =====================================================================
// Attention: one wave32 per (batch, query i, head). qkv rows = [q|k|v] (3H).
// =====================================================================
__global__ void attn_kernel(const float* __restrict__ qkv, const int* __restrict__ lens,
                            float* __restrict__ out, int S, int totalWaves)
{
  extern __shared__ float smem[];
  const int wvb  = threadIdx.x >> 5;
  const int lane = threadIdx.x & 31;
  const int wave = blockIdx.x * 8 + wvb;
  if (wave >= totalWaves) return;                 // uniform per wave
  float* sc = smem + wvb * S;

  const int h  = wave & 7;
  const int t2 = wave >> 3;
  const int i  = t2 % S;
  const int bb = t2 / S;
  const int len = lens[bb];

  const float* base = qkv + (size_t)bb * S * (3*H_);
  const float qv = base[(size_t)i * (3*H_) + h*32 + lane];

  float mx = -1e30f;
  for (int j = 0; j < len; ++j) {
    float kv = base[(size_t)j * (3*H_) + H_ + h*32 + lane];
    float p = qv * kv;
    #pragma unroll
    for (int off = 16; off > 0; off >>= 1) p += __shfl_xor(p, off, 32);
    p *= SCALE_C;
    if (lane == 0) sc[j] = p;
    mx = fmaxf(mx, p);
  }
  asm volatile("s_wait_dscnt 0" ::: "memory");
  __builtin_amdgcn_wave_barrier();

  float denom = 0.0f;
  for (int j = lane; j < len; j += 32) { float e = __expf(sc[j] - mx); sc[j] = e; denom += e; }
  #pragma unroll
  for (int off = 16; off > 0; off >>= 1) denom += __shfl_xor(denom, off, 32);
  asm volatile("s_wait_dscnt 0" ::: "memory");
  __builtin_amdgcn_wave_barrier();

  const float inv = 1.0f / denom;
  float o = 0.0f;
  for (int j = 0; j < len; ++j) {
    float vv = base[(size_t)j * (3*H_) + 2*H_ + h*32 + lane];
    o += sc[j] * vv;
  }
  out[(size_t)(bb * S + i) * H_ + h*32 + lane] = o * inv;
}

// ---------------- LayerNorm(x + res) row kernel (H=256) ----------------
__global__ void ln_res_kernel(float* __restrict__ x, const float* __restrict__ res,
                              const float* __restrict__ g, const float* __restrict__ b)
{
  __shared__ float sred[H_];
  const int r = blockIdx.x, t = threadIdx.x;
  const size_t base = (size_t)r * H_;
  float v = x[base + t] + res[base + t];
  sred[t] = v; __syncthreads();
  for (int s = H_/2; s > 0; s >>= 1) { if (t < s) sred[t] += sred[t + s]; __syncthreads(); }
  const float mean = sred[0] / (float)H_; __syncthreads();
  const float d = v - mean;
  sred[t] = d * d; __syncthreads();
  for (int s = H_/2; s > 0; s >>= 1) { if (t < s) sred[t] += sred[t + s]; __syncthreads(); }
  const float var = sred[0] / (float)H_;
  x[base + t] = d * rsqrtf(var + 1e-5f) * g[t] + b[t];
}

// ---------------- elementwise helpers ----------------
__global__ void concat_kernel(const float* e, const float* c, float* o, int rows, int ed, int cd)
{
  int idx = blockIdx.x * blockDim.x + threadIdx.x;
  int w = ed + cd;
  if (idx >= rows * w) return;
  int r = idx / w, col = idx % w;
  o[idx] = (col < ed) ? e[(size_t)r * ed + col] : c[(size_t)r * cd + (col - ed)];
}
__global__ void hw_combine_kernel(const float* xin, const float* t, const float* g, float* o, int n)
{
  int i = blockIdx.x * blockDim.x + threadIdx.x;
  if (i >= n) return;
  float gv = g[i];
  o[i] = gv * t[i] + (1.0f - gv) * xin[i];
}
__global__ void mulvec_kernel(const float* x, const float* w, float* o, int rows, int width)
{
  int i = blockIdx.x * blockDim.x + threadIdx.x;
  if (i >= rows * width) return;
  o[i] = x[i] * w[i % width];
}
__global__ void rowdot_kernel(const float* X, const float* w, float* o, int rows, int width)
{
  int r = blockIdx.x * blockDim.x + threadIdx.x;
  if (r >= rows) return;
  const float* p = X + (size_t)r * width;
  float s = 0.0f;
  for (int i = 0; i < width; ++i) s += p[i] * w[i];
  o[r] = s;
}
__global__ void u_add_kernel(float* U, const float* rp, const float* rq, int n)
{
  int idx = blockIdx.x * blockDim.x + threadIdx.x;
  if (idx >= n) return;
  int q = idx % QLEN_;
  int rest = idx / QLEN_;
  int p = rest % PLEN_;
  int bk = rest / PLEN_;
  int b = bk / NP_;
  U[idx] += rp[bk * PLEN_ + p] + rq[b * QLEN_ + q];
}
__global__ void softmax_rows_kernel(const float* X, float* O, int rows, int w)
{
  int r = blockIdx.x * blockDim.x + threadIdx.x;
  if (r >= rows) return;
  const float* p = X + (size_t)r * w;
  float* o = O + (size_t)r * w;
  float mx = -1e30f;
  for (int i = 0; i < w; ++i) mx = fmaxf(mx, p[i]);
  float s = 0.0f;
  for (int i = 0; i < w; ++i) { float e = __expf(p[i] - mx); o[i] = e; s += e; }
  float inv = 1.0f / s;
  for (int i = 0; i < w; ++i) o[i] *= inv;
}
__global__ void softmax_strided_kernel(const float* X, float* O, int cols, int n, int stride)
{
  int c = blockIdx.x * blockDim.x + threadIdx.x;
  if (c >= cols) return;
  int bk = c / QLEN_, q = c % QLEN_;
  const float* p = X + (size_t)bk * (PLEN_*QLEN_) + q;
  float* o = O + (size_t)bk * (PLEN_*QLEN_) + q;
  float mx = -1e30f;
  for (int i = 0; i < n; ++i) mx = fmaxf(mx, p[(size_t)i * stride]);
  float s = 0.0f;
  for (int i = 0; i < n; ++i) { float e = __expf(p[(size_t)i * stride] - mx); o[(size_t)i * stride] = e; s += e; }
  float inv = 1.0f / s;
  for (int i = 0; i < n; ++i) o[(size_t)i * stride] *= inv;
}
__global__ void maxk_kernel(const float* Bx, float* Bm, int n)   // [b,k,q,h] -> max over k
{
  int t = blockIdx.x * blockDim.x + threadIdx.x;
  if (t >= n) return;                 // n = B*QLEN*H
  int b = t / (QLEN_ * H_);
  int rem = t % (QLEN_ * H_);
  float m = -1e30f;
  for (int k = 0; k < NP_; ++k)
    m = fmaxf(m, Bx[(size_t)(b * NP_ + k) * (QLEN_ * H_) + rem]);
  Bm[t] = m;
}
__global__ void gp_build_kernel(const float* Ep, const float* A1, const float* A2, float* Gp, int n)
{
  int idx = blockIdx.x * blockDim.x + threadIdx.x;
  if (idx >= n) return;               // n = RP_*5H
  int row = idx / (5 * H_);
  int c = idx % (5 * H_);
  int chunk = c / H_, h = c % H_;
  size_t off = (size_t)row * H_ + h;
  float e = Ep[off], a1 = A1[off], a2 = A2[off];
  float v;
  if      (chunk == 0) v = e;
  else if (chunk == 1) v = a1;
  else if (chunk == 2) v = a2;
  else if (chunk == 3) v = e * a1;
  else                 v = e * a2;
  Gp[idx] = v;
}
__global__ void gq_build_kernel(const float* Eq, const float* B1m, const float* B2m, float* Gq, int n)
{
  int idx = blockIdx.x * blockDim.x + threadIdx.x;
  if (idx >= n) return;               // n = RQ_*5H
  int row = idx / (5 * H_);
  int c = idx % (5 * H_);
  int chunk = c / H_, h = c % H_;
  size_t off = (size_t)row * H_ + h;
  float e = Eq[off], b1 = B1m[off], b2 = B2m[off];
  float v;
  if      (chunk == 0) v = e;
  else if (chunk == 1) v = b1;
  else if (chunk == 2) v = b2;
  else if (chunk == 3) v = e * b1;
  else                 v = e * b2;
  Gq[idx] = v;
}
__global__ void heads_kernel(const float* Mp, const float* ranker, const float* anspos,
                             float* beta, float* prob)
{
  int t = threadIdx.x;
  if (t < B_ * NP_) {
    const float* row = Mp + (size_t)t * PLEN_ * H_;   // p == 0 row
    float s = 0.0f;
    for (int h = 0; h < H_; ++h) s += row[h] * ranker[h];
    beta[t] = sigmoidf_(s);
  } else if (t < B_ * NP_ + B_) {
    int b = t - B_ * NP_;
    float s = 0.0f;
    for (int k = 0; k < NP_; ++k) {
      const float* row = Mp + (size_t)(b * NP_ + k) * PLEN_ * H_;
      for (int h = 0; h < H_; ++h) s += row[h] * anspos[k * H_ + h];
    }
    prob[b] = sigmoidf_(s);
  }
}
__global__ void qmask_kernel(const int* ql, float* o)
{
  int t = blockIdx.x * blockDim.x + threadIdx.x;
  if (t >= B_ * QLEN_) return;
  o[t] = ((t % QLEN_) >= ql[t / QLEN_]) ? 1.0f : 0.0f;
}
__global__ void pmask_kernel(const int* sl, float* o)
{
  int t = blockIdx.x * blockDim.x + threadIdx.x;
  if (t >= RP_) return;
  o[t] = ((t % PLEN_) >= sl[t / PLEN_]) ? 1.0f : 0.0f;
}

// ---------------- encoder driver ----------------
struct EncW { const float *aw,*ab,*ow,*ob,*l1g,*l1b,*f1w,*f1b,*f2w,*f2b,*l2g,*l2b; };

static void run_encoder(float* x, int rows, int nbatch, int S, const int* lens,
                        const EncW& w, float* qkv, float* attn, float* tmp, float* ffh,
                        hipStream_t st)
{
  for (int i = 0; i < 2; ++i) {
    launch_gemm<0,1,0>(x, w.aw + (size_t)i*3*H_*H_, w.ab + (size_t)i*3*H_, qkv,
                       rows, 3*H_, H_, H_, H_, 3*H_, st);
    const int tw = nbatch * S * NH_;
    attn_kernel<<<(tw + 7) / 8, 256, (size_t)8 * S * sizeof(float), st>>>(qkv, lens, attn, S, tw);
    launch_gemm<0,1,0>(attn, w.ow + (size_t)i*H_*H_, w.ob + (size_t)i*H_, tmp,
                       rows, H_, H_, H_, H_, H_, st);
    ln_res_kernel<<<rows, H_, 0, st>>>(x, tmp, w.l1g + (size_t)i*H_, w.l1b + (size_t)i*H_);
    launch_gemm<0,1,3>(x, w.f1w + (size_t)i*DFF_*H_, w.f1b + (size_t)i*DFF_, ffh,
                       rows, DFF_, H_, H_, H_, DFF_, st);
    launch_gemm<0,1,0>(ffh, w.f2w + (size_t)i*H_*DFF_, w.f2b + (size_t)i*H_, tmp,
                       rows, H_, DFF_, DFF_, DFF_, H_, st);
    ln_res_kernel<<<rows, H_, 0, st>>>(x, tmp, w.l2g + (size_t)i*H_, w.l2b + (size_t)i*H_);
  }
}

#define EW(n) <<<((n) + 255) / 256, 256, 0, stream>>>

extern "C" void kernel_launch(void* const* d_in, const int* in_sizes, int n_in,
                              void* d_out, int out_size, void* d_ws, size_t ws_size,
                              hipStream_t stream)
{
  (void)in_sizes; (void)n_in; (void)out_size; (void)ws_size;
  const float* e_q  = (const float*)d_in[0];
  const float* c_q  = (const float*)d_in[1];
  const float* e_p  = (const float*)d_in[2];
  const float* c_p  = (const float*)d_in[3];
  const int*   qln  = (const int*)d_in[4];
  const int*   pln  = (const int*)d_in[5];
  const float* t1w  = (const float*)d_in[6];  const float* t1b = (const float*)d_in[7];
  const float* g1w  = (const float*)d_in[8];  const float* g1b = (const float*)d_in[9];
  const float* t2w  = (const float*)d_in[10]; const float* t2b = (const float*)d_in[11];
  const float* g2w  = (const float*)d_in[12]; const float* g2b = (const float*)d_in[13];
  const float* slw  = (const float*)d_in[14]; const float* slb = (const float*)d_in[15];
  EncW encs = { (const float*)d_in[16],(const float*)d_in[17],(const float*)d_in[18],(const float*)d_in[19],
                (const float*)d_in[20],(const float*)d_in[21],(const float*)d_in[22],(const float*)d_in[23],
                (const float*)d_in[24],(const float*)d_in[25],(const float*)d_in[26],(const float*)d_in[27] };
  EncW encq = { (const float*)d_in[28],(const float*)d_in[29],(const float*)d_in[30],(const float*)d_in[31],
                (const float*)d_in[32],(const float*)d_in[33],(const float*)d_in[34],(const float*)d_in[35],
                (const float*)d_in[36],(const float*)d_in[37],(const float*)d_in[38],(const float*)d_in[39] };
  EncW encp = { (const float*)d_in[40],(const float*)d_in[41],(const float*)d_in[42],(const float*)d_in[43],
                (const float*)d_in[44],(const float*)d_in[45],(const float*)d_in[46],(const float*)d_in[47],
                (const float*)d_in[48],(const float*)d_in[49],(const float*)d_in[50],(const float*)d_in[51] };
  const float* dual = (const float*)d_in[52];
  const float* mqw  = (const float*)d_in[53]; const float* mqb = (const float*)d_in[54];
  const float* mpw  = (const float*)d_in[55]; const float* mpb = (const float*)d_in[56];
  const float* rkw  = (const float*)d_in[57]; const float* apw = (const float*)d_in[58];

  float* W   = (float*)d_ws;
  float* out = (float*)d_out;

  // workspace layout (floats)
  const size_t OFF_XP  = 0;                         // RP*1024 : x_p / hp / later Gp
  const size_t OFF_TB  = OFF_XP  + (size_t)RP_*EMB_;
  const size_t OFF_GB  = OFF_TB  + (size_t)RP_*EMB_;
  const size_t OFF_H1  = OFF_GB  + (size_t)RP_*EMB_;   // Ep + EpWx
  const size_t OFF_XQ  = OFF_H1  + (size_t)RP_*EMB_;
  const size_t OFF_TQ  = OFF_XQ  + (size_t)RQ_*EMB_;
  const size_t OFF_GQB = OFF_TQ  + (size_t)RQ_*EMB_;
  const size_t OFF_H1Q = OFF_GQB + (size_t)RQ_*EMB_;
  const size_t OFF_QKV = OFF_H1Q + (size_t)RQ_*EMB_;
  const size_t OFF_ATT = OFF_QKV + (size_t)RP_*3*H_;
  const size_t OFF_TMP = OFF_ATT + (size_t)RP_*H_;
  const size_t OFF_FFH = OFF_TMP + (size_t)RP_*H_;

  float* XP  = W + OFF_XP;   float* TB = W + OFF_TB;  float* GB = W + OFF_GB;
  float* H1  = W + OFF_H1;   float* XQ = W + OFF_XQ;  float* TQ = W + OFF_TQ;
  float* GQB = W + OFF_GQB;  float* H1Q = W + OFF_H1Q;
  float* QKV = W + OFF_QKV;  float* ATT = W + OFF_ATT;
  float* TMP = W + OFF_TMP;  float* FFH = W + OFF_FFH;

  float* EP   = H1;                          // RP*256
  float* EPWX = H1 + (size_t)RP_*H_;         // RP*256
  float* EQ   = H1Q;                         // RQ*256
  // GB sub-buffers (reused after highway)
  float* U    = GB;                          // 800000
  float* ASMX = GB + 800000;                 // 800000
  float* BSMX = GB + 1600000;                // 800000
  float* RPV  = GB + 2400000;                // 16000
  float* RQV  = GB + 2420000;                // 200
  float* A1B  = GB + 2500000;                // RP*256
  float* A2B  = GB + 6596000;                // RP*256
  float* B1B  = GB + 10692000;               // B*NP*QLEN*H = 512000
  float* B2B  = GB + 11204000;               // 512000
  float* B1M  = GB + 11716000;               // RQ*256
  float* B2M  = GB + 11767200;               // RQ*256
  float* GP   = W + OFF_XP;                  // RP*1280 (overlays XP+TB, both dead)
  float* GQ   = W + OFF_XP + (size_t)RP_*5*H_;  // RQ*1280

  // d_out layout (floats)
  const size_t O_BETA = 0;
  const size_t O_PROB = O_BETA + B_*NP_;
  const size_t O_MQ   = O_PROB + B_;
  const size_t O_QM   = O_MQ   + (size_t)RQ_*H_;
  const size_t O_MP   = O_QM   + (size_t)B_*QLEN_;
  const size_t O_PM   = O_MP   + (size_t)RP_*H_;

  // ---------- 1. concat inputs ----------
  concat_kernel EW(RP_*EMB_)(e_p, c_p, XP, RP_, 256, 768);
  concat_kernel EW(RQ_*EMB_)(e_q, c_q, XQ, RQ_, 256, 768);

  // ---------- 2. highway (p then q) ----------
  launch_gemm<0,1,1>(XP, t1w, t1b, TB, RP_, EMB_, EMB_, EMB_, EMB_, EMB_, stream);
  launch_gemm<0,1,2>(XP, g1w, g1b, GB, RP_, EMB_, EMB_, EMB_, EMB_, EMB_, stream);
  hw_combine_kernel EW(RP_*EMB_)(XP, TB, GB, H1, RP_*EMB_);
  launch_gemm<0,1,1>(H1, t2w, t2b, TB, RP_, EMB_, EMB_, EMB_, EMB_, EMB_, stream);
  launch_gemm<0,1,2>(H1, g2w, g2b, GB, RP_, EMB_, EMB_, EMB_, EMB_, EMB_, stream);
  hw_combine_kernel EW(RP_*EMB_)(H1, TB, GB, XP, RP_*EMB_);

  launch_gemm<0,1,1>(XQ, t1w, t1b, TQ, RQ_, EMB_, EMB_, EMB_, EMB_, EMB_, stream);
  launch_gemm<0,1,2>(XQ, g1w, g1b, GQB, RQ_, EMB_, EMB_, EMB_, EMB_, EMB_, stream);
  hw_combine_kernel EW(RQ_*EMB_)(XQ, TQ, GQB, H1Q, RQ_*EMB_);
  launch_gemm<0,1,1>(H1Q, t2w, t2b, TQ, RQ_, EMB_, EMB_, EMB_, EMB_, EMB_, stream);
  launch_gemm<0,1,2>(H1Q, g2w, g2b, GQB, RQ_, EMB_, EMB_, EMB_, EMB_, EMB_, stream);
  hw_combine_kernel EW(RQ_*EMB_)(H1Q, TQ, GQB, XQ, RQ_*EMB_);

  // ---------- 3. shared projection + shared encoder ----------
  launch_gemm<0,1,0>(XP, slw, slb, EP, RP_, H_, EMB_, EMB_, EMB_, H_, stream);
  launch_gemm<0,1,0>(XQ, slw, slb, EQ, RQ_, H_, EMB_, EMB_, EMB_, H_, stream);
  run_encoder(EP, RP_, B_*NP_, PLEN_, pln, encs, QKV, ATT, TMP, FFH, stream);
  run_encoder(EQ, RQ_, B_,     QLEN_, qln, encs, QKV, ATT, TMP, FFH, stream);

  // ---------- 4. dual attention ----------
  mulvec_kernel EW(RP_*H_)(EP, dual + 2*H_, EPWX, RP_, H_);
  rowdot_kernel EW(RP_)(EP, dual,        RPV, RP_, H_);
  rowdot_kernel EW(RQ_)(EQ, dual + H_,   RQV, RQ_, H_);
  // U (batched over B): (NP*PLEN x H) @ Eq^T
  launch_gemm<0,1,0>(EPWX, EQ, nullptr, U,
                     NP_*PLEN_, QLEN_, H_, H_, H_, QLEN_, stream,
                     B_, (size_t)NP_*PLEN_*H_, (size_t)QLEN_*H_, (size_t)NP_*PLEN_*QLEN_, 1);
  u_add_kernel EW(RP_*QLEN_)(U, RPV, RQV, RP_*QLEN_);
  softmax_rows_kernel EW(RP_)(U, ASMX, RP_, QLEN_);
  softmax_strided_kernel EW(B_*NP_*QLEN_)(U, BSMX, B_*NP_*QLEN_, PLEN_, QLEN_);

  // A1[p,h] = Asm(400x50) @ Eq(50x256)           (batched over bk, Eq shared per b)
  launch_gemm<0,0,0>(ASMX, EQ, nullptr, A1B,
                     PLEN_, H_, QLEN_, QLEN_, H_, H_, stream,
                     B_*NP_, (size_t)PLEN_*QLEN_, (size_t)QLEN_*H_, (size_t)PLEN_*H_, NP_);
  // B1[q,h] = Bsm^T(50x400) @ Ep(400x256)
  launch_gemm<1,0,0>(BSMX, EP, nullptr, B1B,
                     QLEN_, H_, PLEN_, QLEN_, H_, H_, stream,
                     B_*NP_, (size_t)PLEN_*QLEN_, (size_t)PLEN_*H_, (size_t)QLEN_*H_, 1);
  // A2[p,h] = Asm(400x50) @ B1(50x256)
  launch_gemm<0,0,0>(ASMX, B1B, nullptr, A2B,
                     PLEN_, H_, QLEN_, QLEN_, H_, H_, stream,
                     B_*NP_, (size_t)PLEN_*QLEN_, (size_t)QLEN_*H_, (size_t)PLEN_*H_, 1);
  // B2[q,h] = Bsm^T(50x400) @ A1(400x256)
  launch_gemm<1,0,0>(BSMX, A1B, nullptr, B2B,
                     QLEN_, H_, PLEN_, QLEN_, H_, H_, stream,
                     B_*NP_, (size_t)PLEN_*QLEN_, (size_t)PLEN_*H_, (size_t)QLEN_*H_, 1);

  maxk_kernel EW(RQ_*H_)(B1B, B1M, RQ_*H_);
  maxk_kernel EW(RQ_*H_)(B2B, B2M, RQ_*H_);

  // ---------- 5. G matrices + modeling projections ----------
  gp_build_kernel EW(RP_*5*H_)(EP, A1B, A2B, GP, RP_*5*H_);
  gq_build_kernel EW(RQ_*5*H_)(EQ, B1M, B2M, GQ, RQ_*5*H_);

  float* MPX = out + O_MP;   // M_p computed in place in d_out
  float* MQX = out + O_MQ;   // M_q computed in place in d_out
  launch_gemm<0,1,0>(GP, mpw, mpb, MPX, RP_, H_, 5*H_, 5*H_, 5*H_, H_, stream);
  launch_gemm<0,1,0>(GQ, mqw, mqb, MQX, RQ_, H_, 5*H_, 5*H_, 5*H_, H_, stream);
  run_encoder(MPX, RP_, B_*NP_, PLEN_, pln, encp, QKV, ATT, TMP, FFH, stream);
  run_encoder(MQX, RQ_, B_,     QLEN_, qln, encq, QKV, ATT, TMP, FFH, stream);

  // ---------- 6. heads + masks ----------
  heads_kernel<<<1, 64, 0, stream>>>(MPX, rkw, apw, out + O_BETA, out + O_PROB);
  qmask_kernel EW(B_*QLEN_)(qln, out + O_QM);
  pmask_kernel EW(RP_)(pln, out + O_PM);
}
